// H2GCNConv_35588099015572
// MI455X (gfx1250) — compile-verified, matrix operand order checked by reading, not run on hardware
//
#include <hip/hip_runtime.h>
#include <hip/hip_bf16.h>

typedef _Float16 v16h __attribute__((ext_vector_type(16)));
typedef _Float16 v8h  __attribute__((ext_vector_type(8)));
typedef float    v8f  __attribute__((ext_vector_type(8)));
typedef float    v4f  __attribute__((ext_vector_type(4)));

#define NN    8192
#define DD    256
#define KBLK  32
#define NKB   (NN / KBLK)      // 256 fragment k-blocks
#define KSTEP 64               // K staged per barrier (2 fragment k-blocks)
#define NIT   (NN / KSTEP)     // 128 main-loop iterations

// ---------------------------------------------------------------------------
// Pre-pass: pack X (fp32, row-major [K=8192, D=256]) into f16 WMMA B-fragment
// layout in d_ws (4 MB, stays L2-resident).  Fragment element index:
//   Bf[ ((kb*16 + ng)*32 + lane)*16 + e ]
// lane = h*16 + n ; e<8 -> K = kb*32 + h*8 + e ; e>=8 -> K = kb*32 + 16 + h*8 + (e-8)
// Each lane's 16 f16 (32 B) are contiguous -> two coalesced b128 loads later.
// ---------------------------------------------------------------------------
__global__ __launch_bounds__(256) void xpack_kernel(const float* __restrict__ X,
                                                    _Float16* __restrict__ Bf) {
    int bx   = blockIdx.x;        // 0..4095 = kb*16 + ng
    int kb   = bx >> 4;
    int ng   = bx & 15;
    int t    = threadIdx.x;       // 0..255
    int lane = t >> 3;            // 0..31
    int e    = (t & 7) * 2;       // even element index (pairs)
    int h    = lane >> 4;
    int n    = lane & 15;
    int k    = kb * KBLK + ((e < 8) ? (h * 8 + e) : (16 + h * 8 + (e - 8)));
    int col  = ng * 16 + n;

    float f0 = X[(size_t)k * DD + col];
    float f1 = X[(size_t)(k + 1) * DD + col];

    union { _Float16 hh[2]; unsigned u; } pk;
    pk.hh[0] = (_Float16)f0;
    pk.hh[1] = (_Float16)f1;
    _Float16* dst = Bf + (((size_t)(kb * 16 + ng) * 32 + lane) * 16 + e);
    *(unsigned*)dst = pk.u;       // 4B-aligned packed store
}

// ---------------------------------------------------------------------------
// Main kernel: C[tile_m:tile_m+64, :] = A @ X for A = adj_t (z=0) or adj_t2 (z=1).
// WG = 256 threads = 8 waves; WG tile 64(M) x 256(N) -> each adjacency byte is
// read from HBM exactly once (non-temporal: don't pollute L2).
// Waves arranged 2(M) x 4(N); wave tile 32 x 64 -> acc[2][4] of v8f.
// Per iteration a 64x64 fp32 A-slab is staged: global NT load -> cvt to f16
// fragment layout -> LDS (double buffered, 8 KB/buf), one barrier per 16 WMMAs.
// ---------------------------------------------------------------------------
__global__ __launch_bounds__(256) void h2gcn_wmma_kernel(
        const float* __restrict__ A0, const float* __restrict__ A1,
        const _Float16* __restrict__ Bf, float* __restrict__ out) {
    // per buffer: 2 sub-blocks * (4 row-groups * 32 lanes * 16 f16) = 8 KB
    __shared__ __align__(32) _Float16 ldsA[2][4096];

    const float* A = (blockIdx.z == 0) ? A0 : A1;
    const int tile_m = blockIdx.x * 64;
    const int t      = threadIdx.x;
    const int lane   = t & 31;
    const int w      = t >> 5;
    const int wave_m   = (w >> 2) * 32;   // 0 or 32
    const int wave_ng0 = (w & 3) * 4;     // n-group base: 0,4,8,12

    // --- staging assignment: thread t converts 8 consecutive fp32 of one row,
    //     for each of the two 32-wide K sub-blocks of the 64-wide slab ---
    const int sr  = t >> 2;               // row in tile: 0..63
    const int sc8 = (t & 3) * 8;          // k offset within 32-block: 0,8,16,24
    const int sh  = (sc8 >> 3) & 1;       // half (lane group)
    const int se0 = (sc8 >> 4) << 3;      // element base 0 or 8
    const int smg = sr >> 4;              // row group 0..3
    const int sm  = sr & 15;
    const v4f* gsrc_base = (const v4f*)(A + (size_t)(tile_m + sr) * NN + sc8);
    const int ldsoff = (smg * 32 + sh * 16 + sm) * 16 + se0; // f16 elements

    auto stage = [&](int it, int buf) {
#pragma unroll
        for (int sub = 0; sub < 2; ++sub) {
            // it*64 + sub*32 floats = it*16 + sub*8 v4f's
            const v4f* p = gsrc_base + (size_t)it * 16 + sub * 8;
            v4f f0 = __builtin_nontemporal_load(p);       // TH=NT: stream A
            v4f f1 = __builtin_nontemporal_load(p + 1);
            v8h hv;
            hv[0] = (_Float16)f0.x; hv[1] = (_Float16)f0.y;
            hv[2] = (_Float16)f0.z; hv[3] = (_Float16)f0.w;
            hv[4] = (_Float16)f1.x; hv[5] = (_Float16)f1.y;
            hv[6] = (_Float16)f1.z; hv[7] = (_Float16)f1.w;
            *(v8h*)&ldsA[buf][sub * 2048 + ldsoff] = hv;  // one ds_store_b128
        }
    };

    v8f acc[2][4] = {};

    stage(0, 0);
    __syncthreads();

    for (int it = 0; it < NIT; ++it) {
        const int cur = it & 1;
        if (it + 1 < NIT) stage(it + 1, cur ^ 1);   // prefetch next 64-K slab

#pragma unroll
        for (int sub = 0; sub < 2; ++sub) {
            const int kb = it * 2 + sub;            // fragment k-block index
            v16h afrag[2];
#pragma unroll
            for (int i = 0; i < 2; ++i) {
                int mg = (wave_m >> 4) + i;
                afrag[i] = *(const v16h*)&ldsA[cur][sub * 2048 + (mg * 32 + lane) * 16];
            }
#pragma unroll
            for (int j = 0; j < 4; ++j) {
                int ng = wave_ng0 + j;
                v16h bfrag = *(const v16h*)(
                    Bf + (((size_t)(kb * 16 + ng) * 32 + lane) * 16));
#pragma unroll
                for (int i = 0; i < 2; ++i) {
                    acc[i][j] = __builtin_amdgcn_wmma_f32_16x16x32_f16(
                        false, afrag[i], false, bfrag,
                        (short)0, acc[i][j], false, false);
                }
            }
        }
        __syncthreads();
    }

    // --- epilogue: C/D layout -> lanes 0-15: M=r, lanes 16-31: M=8+r.
    //     Output is write-once: non-temporal stores, keep L2 for packed X. ---
    const int h = lane >> 4;
    const int n = lane & 15;
    const size_t colbase = (size_t)blockIdx.z * DD + (size_t)(w & 3) * 64 + n;
#pragma unroll
    for (int i = 0; i < 2; ++i) {
        const int rowb = tile_m + wave_m + i * 16 + h * 8;
#pragma unroll
        for (int j = 0; j < 4; ++j) {
            const size_t col = colbase + (size_t)j * 16;
#pragma unroll
            for (int r = 0; r < 8; ++r) {
                __builtin_nontemporal_store(
                    acc[i][j][r], out + (size_t)(rowb + r) * (2 * DD) + col);
            }
        }
    }
}

extern "C" void kernel_launch(void* const* d_in, const int* in_sizes, int n_in,
                              void* d_out, int out_size, void* d_ws, size_t ws_size,
                              hipStream_t stream) {
    const float* x    = (const float*)d_in[0];   // [8192, 256]
    const float* adj  = (const float*)d_in[1];   // [8192, 8192]
    const float* adj2 = (const float*)d_in[2];   // [8192, 8192]
    float* out        = (float*)d_out;           // [8192, 512]
    _Float16* Bf      = (_Float16*)d_ws;         // 4 MB packed f16 X

    xpack_kernel<<<dim3(NKB * 16), 256, 0, stream>>>(x, Bf);
    h2gcn_wmma_kernel<<<dim3(NN / 64, 1, 2), 256, 0, stream>>>(adj, adj2, Bf, out);
}
